// CloudNeRF_46969762349679
// MI455X (gfx1250) — matrix operand order repaired and checked by reading, not compiled, not run on hardware
//
#include <hip/hip_runtime.h>

#define NPTS   65536
#define NC     2048
#define KNN    16
#define NPAIRS (NPTS/32)     // 32 rows (two 16-row M-tiles) per wave

typedef __attribute__((ext_vector_type(16))) _Float16 v16h;
typedef __attribute__((ext_vector_type(8)))  _Float16 v8h;
typedef __attribute__((ext_vector_type(8)))  float    v8f;

union AFrag { v16h v; v8h h[2]; };

// ---- packed-weight offsets (halves, in the global packed buffer) ----
#define OFF_W0 0
#define OFF_W1 16384
#define OFF_W2 32768      // 256x128
#define OFF_W3 65536
#define OFF_WF 81920
#define OFF_WD 98304      // 160x64
#define OFF_WS 108544     // 128x16
#define OFF_WR 110592     // 64x16
#define W_HALVES 111616

// Only W0 + W2 staged in LDS (hot, LDS-bandwidth-friendly); the rest ride VMEM/L2.
#define OFF_LW0 0
#define OFF_LW2 16384
#define LDSW_HALVES 49152
#define ACT_HALVES_PER_WAVE 8192      // four 16x128 f16 buffers
#define MLP_WAVES 8
#define SMEM_HALVES (LDSW_HALVES + MLP_WAVES*ACT_HALVES_PER_WAVE) // 114688 h = 229376 B

// =====================================================================
// Weight packer: f32 row-major (Ksrc x Nsrc) -> f16 B-fragment layout
// dst[( (t*Npad + n) * 32 ) + 16*hi + j] = W[K = 32t+16hi+j][n]
// biasKK: K-slot that receives the bias row (ones-trick), -1 = none.
// =====================================================================
__global__ void pack_weights(const float* __restrict__ src, _Float16* __restrict__ dst,
                             int Nsrc, int Npad, int Kpad,
                             int c0, int s1dst, int s1src, int s1cnt,
                             const float* __restrict__ brow, int biasKK) {
  int i = blockIdx.x * blockDim.x + threadIdx.x;
  int total = Kpad * Npad;
  if (i >= total) return;
  int j    = i & 15;
  int hi   = (i >> 4) & 1;
  int rest = i >> 5;            // t*Npad + n
  int n    = rest % Npad;
  int t    = rest / Npad;
  int kk   = t * 32 + hi * 16 + j;
  float v = 0.f;
  int row = -1;
  if (kk < c0) row = kk;
  else if (s1cnt > 0 && kk >= s1dst && kk < s1dst + s1cnt) row = s1src + (kk - s1dst);
  if (row >= 0 && n < Nsrc) v = src[(size_t)row * Nsrc + n];
  if (kk == biasKK && n < Nsrc) v = brow[n];
  dst[i] = (_Float16)v;
}

// =====================================================================
// One-time: bitonic-sort the 2048 codes by x (u64 key = x_bits<<32 | idx).
// Emits sortedP (x,y,z,|c|^2) and sortedId (original index).
// =====================================================================
__global__ __launch_bounds__(256)
void sort_codes(const float* __restrict__ cpos,
                float4* __restrict__ sortedP, unsigned* __restrict__ sortedId) {
  __shared__ unsigned long long key[NC];
  for (int i = threadIdx.x; i < NC; i += 256) {
    float x = cpos[i * 3];                 // x in [0,1) -> bits are monotone
    key[i] = ((unsigned long long)__float_as_uint(x) << 32) | (unsigned)i;
  }
  __syncthreads();
  for (int k = 2; k <= NC; k <<= 1) {
    for (int j = k >> 1; j > 0; j >>= 1) {
      for (int t = threadIdx.x; t < NC; t += 256) {
        int ixj = t ^ j;
        if (ixj > t) {
          bool up = ((t & k) == 0);
          unsigned long long a = key[t], b = key[ixj];
          bool sw = up ? (a > b) : (a < b);
          if (sw) { key[t] = b; key[ixj] = a; }
        }
      }
      __syncthreads();
    }
  }
  for (int i = threadIdx.x; i < NC; i += 256) {
    unsigned c = (unsigned)(key[i] & 0xFFFFFFFFu);
    float x = cpos[c*3+0], y = cpos[c*3+1], z = cpos[c*3+2];
    sortedP[i] = make_float4(x, y, z, x*x + y*y + z*z);
    sortedId[i] = c;
  }
}

// =====================================================================
// KNN(16) with x-sorted window scan: binary-search qx, expand two cursors
// outward (nearest-x first), stop when dx^2 exceeds the 16th-NN bound.
// Keys: monotone-u32 distance with the SORTED position in the low 11 bits;
// sorted-insert is pure v_min_u32/v_max_u32. Exact distances recomputed.
// =====================================================================
__global__ __launch_bounds__(256)
void knn_pack(const float* __restrict__ qp,        // N x 3
              const float* __restrict__ xyzdir,    // N x 90
              const float4* __restrict__ gsP,      // C, sorted by x
              const unsigned* __restrict__ gsId,   // C
              const float* __restrict__ codes,     // C x 64
              _Float16* __restrict__ X,            // N x 128
              _Float16* __restrict__ Dir) {        // N x 32
  __shared__ float4   sP[NC];
  __shared__ unsigned sId[NC];
  for (int i = threadIdx.x; i < NC; i += blockDim.x) {
    sP[i] = gsP[i];
    sId[i] = gsId[i];
  }
  __syncthreads();

  for (int n = blockIdx.x * blockDim.x + threadIdx.x; n < NPTS;
       n += gridDim.x * blockDim.x) {
    float qx = qp[n*3+0], qy = qp[n*3+1], qz = qp[n*3+2];
    float qq  = qx*qx + qy*qy + qz*qz;
    float qx2 = -2.f*qx, qy2 = -2.f*qy, qz2 = -2.f*qz;

    // binary search: first sorted index with x >= qx
    int lo = 0, hiE = NC;
#pragma unroll 1
    for (int it = 0; it < 11; ++it) {
      int mid = (lo + hiE) >> 1;
      if (sP[mid].x < qx) lo = mid + 1; else hiE = mid;
    }
    int jl = lo - 1, jr = lo;
    float xl = (jl >= 0) ? sP[jl].x : -1e19f;
    float xr = (jr < NC) ? sP[jr].x :  1e19f;

    unsigned bd[KNN];
#pragma unroll
    for (int k = 0; k < KNN; ++k) bd[k] = 0xFF7FFFFFu;   // key(FLT_MAX)
    float thr   = 3.4e38f;        // bound on s = d2 - |q|^2
    float d2thr = 3.4e38f;        // bound on d2

#pragma unroll 1
    for (int it = 0; it <= NC; ++it) {
      float dxl = qx - xl, dxr = xr - qx;
      bool tl = dxl < dxr;
      float dxm = tl ? dxl : dxr;
      if (dxm * dxm > d2thr) break;           // both sides out of range
      int j = tl ? jl : jr;
      float4 p = sP[j];
      float s = fmaf(qx2, p.x, fmaf(qy2, p.y, fmaf(qz2, p.z, p.w)));
      if (s <= thr) {
        unsigned ub   = __float_as_uint(s);
        unsigned mono = (s < 0.f) ? ~ub : (ub | 0x80000000u);
        unsigned kk   = (mono & 0xFFFFF800u) | (unsigned)j;
#pragma unroll
        for (int k = 0; k < KNN; ++k) {
          unsigned old = bd[k];
          bd[k] = (old < kk) ? old : kk;   // v_min_u32
          kk    = (old < kk) ? kk  : old;  // v_max_u32
        }
        unsigned tu = bd[KNN-1] | 0x7FFu;  // conservative f32 upper bound
        thr = (tu & 0x80000000u) ? __uint_as_float(tu ^ 0x80000000u)
                                 : __uint_as_float(~tu);
        d2thr = thr + qq;
      }
      if (tl) { --jl; xl = (jl >= 0) ? sP[jl].x : -1e19f; }
      else    { ++jr; xr = (jr < NC) ? sP[jr].x :  1e19f; }
    }

    float w[KNN]; int bi[KNN]; float wsum = 0.f;
#pragma unroll
    for (int k = 0; k < KNN; ++k) {
      int j = (int)(bd[k] & 0x7FFu);
      bi[k] = (int)sId[j];
      float4 p = sP[j];
      float dx = qx - p.x, dy = qy - p.y, dz = qz - p.z;
      float sq = dx*dx + dy*dy + dz*dz + 1e-16f;
      w[k] = 1.f / sq;                     // DIST_SCALE=2 -> w = 1/d2
      wsum += w[k];
    }
    float inv = 1.f / wsum;

    float acc[64];
#pragma unroll
    for (int d = 0; d < 64; ++d) acc[d] = 0.f;
    for (int k = 0; k < KNN; ++k) {
      const float4* cr = (const float4*)(codes + (size_t)bi[k] * 64);
      float wk = w[k] * inv;
#pragma unroll
      for (int d4 = 0; d4 < 16; ++d4) {
        float4 v = cr[d4];
        acc[d4*4+0] += wk * v.x; acc[d4*4+1] += wk * v.y;
        acc[d4*4+2] += wk * v.z; acc[d4*4+3] += wk * v.w;
      }
    }

    _Float16* xr16 = X + (size_t)n * 128;
#pragma unroll
    for (int d = 0; d < 64; ++d) xr16[d] = (_Float16)acc[d];
    const float* xe = xyzdir + (size_t)n * 90;
#pragma unroll
    for (int d = 0; d < 63; ++d) xr16[64 + d] = (_Float16)xe[d];
    xr16[127] = (_Float16)1.f;                     // ones-trick: b0/b2 rows
    _Float16* dr = Dir + (size_t)n * 32;
#pragma unroll
    for (int d = 0; d < 27; ++d) dr[d] = (_Float16)xe[63 + d];
    dr[27] = (_Float16)1.f;                        // ones-trick: bd row
#pragma unroll
    for (int d = 28; d < 32; ++d) dr[d] = (_Float16)0.f;
  }
}

// =====================================================================
// WMMA MLP helpers
// =====================================================================
__device__ __forceinline__ v8f vzero() {
  v8f z = {0.f,0.f,0.f,0.f,0.f,0.f,0.f,0.f}; return z;
}

// A fragment (16x32 f16): row = lane&15; lanes 0-15: K k0..k0+15,
// lanes 16-31: K k0+16..31  (contiguous, matches packed B)
__device__ __forceinline__ v16h load_a(const _Float16* row, int k0, int hi) {
  AFrag f;
  f.h[0] = *(const v8h*)(row + k0 + hi*16);
  f.h[1] = *(const v8h*)(row + k0 + hi*16 + 8);
  return f.v;
}

// B fragment (packed layout, LDS or global): col n = lane&15, K = 32t+16hi+j
__device__ __forceinline__ v16h load_b(const _Float16* w, int Npad, int t, int n0, int lane) {
  int n = n0 + (lane & 15);
  const _Float16* p = w + (((size_t)(t * Npad + n)) << 5) + ((lane >> 4) << 4);
  AFrag f;
  f.h[0] = *(const v8h*)p;
  f.h[1] = *(const v8h*)(p + 8);
  return f.v;
}

__device__ __forceinline__ v8f wmma_f16(v16h a, v16h b, v8f c) {
  return __builtin_amdgcn_wmma_f32_16x16x32_f16(false, a, false, b, (short)0, c,
                                                false, false);
}

// D layout: col = lane&15, VGPR r -> row r + 8*(lane>>4). bias==nullptr: folded.
__device__ __forceinline__ void store_act(_Float16* buf, int S, int n0, v8f acc,
                                          const float* __restrict__ bias, bool relu,
                                          int lane) {
  int n = n0 + (lane & 15);
  int hi = lane >> 4;
  float b = bias ? bias[n] : 0.f;
#pragma unroll
  for (int r = 0; r < 8; ++r) {
    float v = acc[r];
    if (bias) v += b;
    if (relu) v = fmaxf(v, 0.f);
    buf[(r + 8*hi) * S + n] = (_Float16)v;
  }
}

// Two M-tiles share every B fragment (2 independent WMMAs per B load).
template<int KT, int NT>
__device__ __forceinline__ void gemm2(const v16h* a0, const v16h* a1,
                                      const _Float16* w, int Npad,
                                      _Float16* d0, _Float16* d1, int S,
                                      const float* __restrict__ bias, bool relu,
                                      int lane) {
#pragma unroll
  for (int nt = 0; nt < NT; ++nt) {
    v8f acc0 = vzero(), acc1 = vzero();
#pragma unroll
    for (int t = 0; t < KT; ++t) {
      v16h b = load_b(w, Npad, t, nt * 16, lane);
      acc0 = wmma_f16(a0[t], b, acc0);
      acc1 = wmma_f16(a1[t], b, acc1);
    }
    store_act(d0, S, nt * 16, acc0, bias, relu, lane);
    store_act(d1, S, nt * 16, acc1, bias, relu, lane);
  }
}

__global__ __launch_bounds__(256)
void mlp_kernel(const _Float16* __restrict__ X, const _Float16* __restrict__ Dir,
                const _Float16* __restrict__ Wp,
                const float* __restrict__ b1, const float* __restrict__ b3,
                const float* __restrict__ bf,
                const float* __restrict__ bs, const float* __restrict__ br,
                float* __restrict__ out) {
  extern __shared__ _Float16 smem[];

  // Stage W0 (32 KB) and W2 (64 KB) into LDS once per workgroup.
  {
    const uint4* g0 = (const uint4*)(Wp + OFF_W0);
    const uint4* g2 = (const uint4*)(Wp + OFF_W2);
    uint4* s0 = (uint4*)(smem + OFF_LW0);
    uint4* s2 = (uint4*)(smem + OFF_LW2);
    for (int i = threadIdx.x; i < 16384 / 8; i += blockDim.x) s0[i] = g0[i];
    for (int i = threadIdx.x; i < 32768 / 8; i += blockDim.x) s2[i] = g2[i];
  }
  __syncthreads();

  const int lane = threadIdx.x & 31;
  const int wave = threadIdx.x >> 5;
  const int l16  = lane & 15;
  const int hi   = lane >> 4;

  _Float16* act   = smem + LDSW_HALVES + wave * ACT_HALVES_PER_WAVE;
  _Float16* bufA0 = act;
  _Float16* bufA1 = act + 2048;
  _Float16* bufB0 = act + 4096;
  _Float16* bufB1 = act + 6144;

  const _Float16* W1G = Wp + OFF_W1;
  const _Float16* W3G = Wp + OFF_W3;
  const _Float16* WfG = Wp + OFF_WF;
  const _Float16* WdG = Wp + OFF_WD;
  const _Float16* WsG = Wp + OFF_WS;
  const _Float16* WrG = Wp + OFF_WR;

  const int wavesTotal = gridDim.x * (blockDim.x >> 5);
  const int waveId     = blockIdx.x * (blockDim.x >> 5) + wave;

  v16h ax0[4], ax1[4];   // X fragments, live across L0..L2 (skip-concat reuse)
  v16h ah0[4], ah1[4];

  for (int pair = waveId; pair < NPAIRS; pair += wavesTotal) {
    const int r0 = pair * 32;
    const _Float16* X0 = X + (size_t)(r0 + l16) * 128;
    const _Float16* X1 = X0 + 16 * 128;

    // prefetch next pair's X rows (global_prefetch_b8, speculative)
    __builtin_prefetch(X0 + (size_t)wavesTotal * 32 * 128, 0, 1);
    __builtin_prefetch(X1 + (size_t)wavesTotal * 32 * 128, 0, 1);

#pragma unroll
    for (int t = 0; t < 4; ++t) {
      ax0[t] = load_a(X0, t * 32, hi);
      ax1[t] = load_a(X1, t * 32, hi);
    }

    // L0: h0 = relu(X @ W0p)      (b0 folded via X[:,127]==1)
    gemm2<4, 8>(ax0, ax1, smem + OFF_LW0, 128, bufA0, bufA1, 128, nullptr, true, lane);

    // L1: h1 = relu(h0 @ W1 + b1)   (B from global)
#pragma unroll
    for (int t = 0; t < 4; ++t) {
      ah0[t] = load_a(bufA0 + l16 * 128, t * 32, hi);
      ah1[t] = load_a(bufA1 + l16 * 128, t * 32, hi);
    }
    gemm2<4, 8>(ah0, ah1, W1G, 128, bufB0, bufB1, 128, b1, true, lane);

    // L2: h2 = relu([xyz | h1] @ W2p)   [K=256, b2 folded]
#pragma unroll
    for (int t = 0; t < 4; ++t) {
      ah0[t] = load_a(bufB0 + l16 * 128, t * 32, hi);
      ah1[t] = load_a(bufB1 + l16 * 128, t * 32, hi);
    }
    {
      v16h c0[8], c1[8];
#pragma unroll
      for (int t = 0; t < 4; ++t) {
        c0[t] = ax0[t]; c0[4 + t] = ah0[t];
        c1[t] = ax1[t]; c1[4 + t] = ah1[t];
      }
      gemm2<8, 8>(c0, c1, smem + OFF_LW2, 128, bufA0, bufA1, 128, nullptr, true, lane);
    }

    // L3: h3 = relu(h2 @ W3 + b3)   (B from global)
#pragma unroll
    for (int t = 0; t < 4; ++t) {
      ah0[t] = load_a(bufA0 + l16 * 128, t * 32, hi);
      ah1[t] = load_a(bufA1 + l16 * 128, t * 32, hi);
    }
    gemm2<4, 8>(ah0, ah1, W3G, 128, bufB0, bufB1, 128, b3, true, lane);

    // h3 fragments for sigma + final
#pragma unroll
    for (int t = 0; t < 4; ++t) {
      ah0[t] = load_a(bufB0 + l16 * 128, t * 32, hi);
      ah1[t] = load_a(bufB1 + l16 * 128, t * 32, hi);
    }

    // sigma = h3 @ Ws + bs   (N padded to 16, col 0 valid; B from global)
    {
      v8f s0 = vzero(), s1 = vzero();
#pragma unroll
      for (int t = 0; t < 4; ++t) {
        v16h b = load_b(WsG, 16, t, 0, lane);
        s0 = wmma_f16(ah0[t], b, s0);
        s1 = wmma_f16(ah1[t], b, s1);
      }
      if (l16 == 0) {
        float bb = bs[0];
#pragma unroll
        for (int r = 0; r < 8; ++r) {
          out[(size_t)(r0      + r + 8*hi) * 4 + 3] = s0[r] + bb;
          out[(size_t)(r0 + 16 + r + 8*hi) * 4 + 3] = s1[r] + bb;
        }
      }
    }

    // final = h3 @ Wf + bf  (no relu; B from global)
    gemm2<4, 8>(ah0, ah1, WfG, 128, bufA0, bufA1, 128, bf, false, lane);

    // d = relu([final | dir] @ Wdp)   [K=160 -> N=64; bd folded; B from global]
    {
      v16h ad0[5], ad1[5];
#pragma unroll
      for (int t = 0; t < 4; ++t) {
        ad0[t] = load_a(bufA0 + l16 * 128, t * 32, hi);
        ad1[t] = load_a(bufA1 + l16 * 128, t * 32, hi);
      }
      ad0[4] = load_a(Dir + (size_t)(r0 + l16) * 32, 0, hi);
      ad1[4] = load_a(Dir + (size_t)(r0 + 16 + l16) * 32, 0, hi);
      gemm2<5, 4>(ad0, ad1, WdG, 64, bufB0, bufB1, 64, nullptr, true, lane);
    }

    // rgb = d @ Wr + br   (N padded to 16, cols 0..2 valid; B from global)
    {
      v16h p0a = load_a(bufB0 + l16 * 64, 0, hi);
      v16h p0b = load_a(bufB0 + l16 * 64, 32, hi);
      v16h p1a = load_a(bufB1 + l16 * 64, 0, hi);
      v16h p1b = load_a(bufB1 + l16 * 64, 32, hi);
      v16h bfr0 = load_b(WrG, 16, 0, 0, lane);
      v16h bfr1 = load_b(WrG, 16, 1, 0, lane);
      v8f s0 = vzero(), s1 = vzero();
      s0 = wmma_f16(p0a, bfr0, s0);
      s1 = wmma_f16(p1a, bfr0, s1);
      s0 = wmma_f16(p0b, bfr1, s0);
      s1 = wmma_f16(p1b, bfr1, s1);
      if (l16 < 3) {
        float bb = br[l16];
#pragma unroll
        for (int r8 = 0; r8 < 8; ++r8) {
          out[(size_t)(r0      + r8 + 8*hi) * 4 + l16] = s0[r8] + bb;
          out[(size_t)(r0 + 16 + r8 + 8*hi) * 4 + l16] = s1[r8] + bb;
        }
      }
    }
  }
}

// =====================================================================
// Launcher
// =====================================================================
extern "C" void kernel_launch(void* const* d_in, const int* in_sizes, int n_in,
                              void* d_out, int out_size, void* d_ws, size_t ws_size,
                              hipStream_t stream) {
  (void)in_sizes; (void)n_in; (void)out_size; (void)ws_size;
  const float* qp    = (const float*)d_in[1];
  const float* xyzd  = (const float*)d_in[2];
  const float* cpos  = (const float*)d_in[3];
  const float* codes = (const float*)d_in[4];
  const float* W0 = (const float*)d_in[5];  const float* b0 = (const float*)d_in[6];
  const float* W1 = (const float*)d_in[7];  const float* b1 = (const float*)d_in[8];
  const float* W2 = (const float*)d_in[9];  const float* b2 = (const float*)d_in[10];
  const float* W3 = (const float*)d_in[11]; const float* b3 = (const float*)d_in[12];
  const float* Wf = (const float*)d_in[13]; const float* bf = (const float*)d_in[14];
  const float* Wd = (const float*)d_in[15]; const float* bd = (const float*)d_in[16];
  const float* Ws = (const float*)d_in[17]; const float* bs = (const float*)d_in[18];
  const float* Wr = (const float*)d_in[19]; const float* br = (const float*)d_in[20];

  _Float16* Wp       = (_Float16*)d_ws;                                // 223232 B
  float4*   sortedP  = (float4*)  ((char*)d_ws + (224 << 10));         // 32 KB
  unsigned* sortedId = (unsigned*)((char*)d_ws + (256 << 10));         // 8 KB
  _Float16* X        = (_Float16*)((char*)d_ws + (320 << 10));         // 16 MB
  _Float16* Dir      = (_Float16*)((char*)d_ws + (320 << 10) + (size_t)NPTS * 128 * 2);

  auto pack = [&](const float* src, int off, int Nsrc, int Npad, int Kpad,
                  int c0, int s1d, int s1s, int s1c,
                  const float* brow, int biasKK) {
    int total = Kpad * Npad;
    pack_weights<<<(total + 255) / 256, 256, 0, stream>>>(
        src, Wp + off, Nsrc, Npad, Kpad, c0, s1d, s1s, s1c, brow, biasKK);
  };
  pack(W0, OFF_W0, 128, 128, 128, 127,   0,   0,   0, b0, 127);  // b0 in K-row 127
  pack(W1, OFF_W1, 128, 128, 128, 128,   0,   0,   0, nullptr, -1);
  pack(W2, OFF_W2, 128, 128, 256, 127, 128, 127, 128, b2, 127);  // b2 in K-row 127
  pack(W3, OFF_W3, 128, 128, 128, 128,   0,   0,   0, nullptr, -1);
  pack(Wf, OFF_WF, 128, 128, 128, 128,   0,   0,   0, nullptr, -1);
  pack(Wd, OFF_WD,  64,  64, 160, 155,   0,   0,   0, bd, 155);  // bd in K-row 155
  pack(Ws, OFF_WS,   1,  16, 128, 128,   0,   0,   0, nullptr, -1);
  pack(Wr, OFF_WR,   3,  16,  64,  64,   0,   0,   0, nullptr, -1);

  sort_codes<<<1, 256, 0, stream>>>(cpos, sortedP, sortedId);

  knn_pack<<<256, 256, 0, stream>>>(qp, xyzd, sortedP, sortedId, codes, X, Dir);

  size_t shmem = (size_t)SMEM_HALVES * sizeof(_Float16);   // 229376 B
  mlp_kernel<<<256, 256, shmem, stream>>>(X, Dir, Wp,
                                          b1, b3, bf, bs, br,
                                          (float*)d_out);
}
